// HorizontalShiftConv2d_47210280517778
// MI455X (gfx1250) — matrix operand order repaired and checked by reading, not compiled, not run on hardware
//
#include <hip/hip_runtime.h>

#define Bb 8
#define Cc 256
#define Hh 96
#define Ww 512
#define Kk 9

typedef __attribute__((ext_vector_type(8)))  float  v8f;
typedef __attribute__((ext_vector_type(8)))  __bf16 v8bf;
typedef __attribute__((ext_vector_type(16))) __bf16 v16bf;

union V16 { v16bf v; v8bf h[2]; };

__device__ __forceinline__ unsigned short f2bf(float f) {
  unsigned int u = __builtin_bit_cast(unsigned int, f);
  unsigned int r = u + 0x7FFFu + ((u >> 16) & 1u);   // round-to-nearest-even
  return (unsigned short)(r >> 16);
}

// ---------------- Phase 0: conv_w fp32 -> bf16 slabs ----------------
// Layout: wprep[cb(4)][cc(8)][k(9)][co(64)][ci(32)], each (cb,cc) slab is a
// contiguous 18432-element (36864 B) block the conv kernel DMAs into LDS.
__global__ __launch_bounds__(256) void hsc_wprep_kernel(const float* __restrict__ conv_w,
                                                        unsigned short* __restrict__ wprep) {
  const int slab = blockIdx.x;                     // 0..31 = cb*8+cc
  const int cb = slab >> 3, cc = slab & 7;
  unsigned short* dst = wprep + (size_t)slab * 18432;
  for (int i = threadIdx.x; i < 18432; i += 256) {
    int k  = i / 2048;
    int r  = i - k * 2048;
    int co = r >> 5;
    int ci = r & 31;
    float wv = conv_w[(((size_t)(cb * 64 + co)) * Cc + (cc * 32 + ci)) * Kk + k];
    dst[i] = f2bf(wv);
  }
}

// ---------------- Phase 1: spatial mean -> sq[b][c] ----------------
__global__ __launch_bounds__(256) void hsc_mean_kernel(const float* __restrict__ x,
                                                       float* __restrict__ sq) {
  const int bc = blockIdx.x;                       // 0 .. B*C-1
  const float4* p = (const float4*)(x + (size_t)bc * (Hh * Ww));
  float s = 0.f;
  for (int i = threadIdx.x; i < (Hh * Ww) / 4; i += 256) {
    float4 v = p[i];
    s += v.x + v.y + v.z + v.w;
  }
  for (int o = 16; o > 0; o >>= 1) s += __shfl_down(s, o, 32);
  __shared__ float ls[8];
  if ((threadIdx.x & 31) == 0) ls[threadIdx.x >> 5] = s;
  __syncthreads();
  if (threadIdx.x == 0) {
    float t = 0.f;
    for (int i = 0; i < 8; ++i) t += ls[i];
    sq[bc] = t * (1.0f / (Hh * Ww));
  }
}

// ------- Phase 2: FC + maxabs-normalize + shift + softmax gate -------
__global__ __launch_bounds__(256) void hsc_gate_kernel(const float* __restrict__ sq,
                                                       const float* __restrict__ fc_w,
                                                       const float* __restrict__ fc_b,
                                                       float* __restrict__ gate,
                                                       int* __restrict__ shift) {
  const int b = blockIdx.x;
  const int c = threadIdx.x;                       // 256 threads = 256 channels
  __shared__ float srow[Cc];
  __shared__ float red[Cc];
  srow[c] = sq[b * Cc + c];
  __syncthreads();
  float acc = fc_b[c];
  const float* wr = fc_w + (size_t)c * Cc;
  for (int i = 0; i < Cc; ++i) acc += srow[i] * wr[i];
  red[c] = fabsf(acc);
  __syncthreads();
  for (int o = 128; o > 0; o >>= 1) {
    if (c < o) red[c] = fmaxf(red[c], red[c + o]);
    __syncthreads();
  }
  const float m = red[0];
  __syncthreads();
  const float hnorm = acc / m;
  const float e = __expf(fabsf(hnorm));
  red[c] = e;
  __syncthreads();
  for (int o = 128; o > 0; o >>= 1) {
    if (c < o) red[c] += red[c + o];
    __syncthreads();
  }
  gate[b * Cc + c] = e / red[0];
  int s = ((int)floorf((float)Ww * hnorm) + Ww) % Ww;
  if (s < 0) s += Ww;
  shift[b * Cc + c] = s;                           // in [0, W)
}

// ------- Phase 3: shifted 1x9 conv as WMMA GEMM + fused epilogue -------
#define XSTR 40            // LDS row stride in halves (32 data + 8 pad)

#if __has_builtin(__builtin_amdgcn_tensor_load_to_lds)
#define HAVE_TDM 1
typedef unsigned int v4u_ __attribute__((ext_vector_type(4)));
typedef int v8i_ __attribute__((ext_vector_type(8)));
typedef int v4i_ __attribute__((ext_vector_type(4)));

// 1-D TDM copy of one 36864-B weight slab into LDS, inserting 4 DWORDs of
// padding after every 16 DWORDs (=> row stride 40 halves, matching XSTR).
__device__ __forceinline__ void tdm_load_weights(unsigned ldsAddr,
                                                 unsigned long long gaddr) {
  v4u_ g0;
  g0[0] = 1u;                                        // count=1, user mode
  g0[1] = ldsAddr;                                   // lds_addr
  g0[2] = (unsigned)(gaddr & 0xFFFFFFFFull);         // global_addr[31:0]
  g0[3] = (unsigned)((gaddr >> 32) & 0x1FFFFFFull)   // global_addr[56:32]
          | (2u << 30);                              // type=2 (image)
  v8i_ g1;
  g1[0] = (int)((1u << 16)      // data_size = 2 bytes
              | (1u << 20)      // pad_enable
              | (3u << 22)      // pad_interval: 16 DWORDs (64 B = one row)
              | (3u << 25));    // pad_amount: 4 DWORDs (8 halves)
  g1[1] = (int)(18432u << 16);  // tensor_dim0[15:0] = 0x4800
  g1[2] = (int)(1u << 16);      // tensor_dim0[31:16]=0, tensor_dim1=1
  g1[3] = (int)(18432u << 16);  // tile_dim0 = 18432 elements
  g1[4] = 0;                    // tile_dim1 = tile_dim2 = 0 (unused)
  g1[5] = 18432;                // tensor_dim0_stride
  g1[6] = 0;
  g1[7] = 0;
  v4i_ z4 = {0, 0, 0, 0};
  v8i_ z8 = {0, 0, 0, 0, 0, 0, 0, 0};
  __builtin_amdgcn_tensor_load_to_lds(g0, g1, z4, z4, z8, 0);
}
#else
#define HAVE_TDM 0
#endif

// Block tile: 64 (cout) x 128 (w) for one (b,h). 8 waves; each wave: one
// 16-row cout slab x four 16-col w tiles. Reduction: 8 cin-chunks x 9 taps,
// each a K=32 bf16 WMMA slice. Weights arrive per chunk via TDM while the
// VALU stages the circular-shifted x window.
__global__ __launch_bounds__(256) void hsc_conv_kernel(const float* __restrict__ x,
                                                       const unsigned short* __restrict__ wprep,
                                                       const int* __restrict__ shift,
                                                       const float* __restrict__ gate,
                                                       float* __restrict__ out) {
  const int tid      = threadIdx.x;
  const int w0       = blockIdx.x * 128;
  const int coutBase = blockIdx.y * 64;
  const int b        = blockIdx.z / Hh;
  const int h        = blockIdx.z % Hh;

  __shared__ unsigned short xT[136 * XSTR];        // [w_local 0..135][cin 0..31]
  __shared__ unsigned short wT[Kk * 64 * XSTR];    // [k][cout 0..63][cin 0..31]
  __shared__ int   sSh[Cc];
  __shared__ float sGate[64];

  sSh[tid] = shift[b * Cc + tid];
  if (tid < 64) sGate[tid] = gate[b * Cc + coutBase + tid];

  const int lane  = tid & 31;
  const int wv    = tid >> 5;
  const int cm    = wv & 3;                        // cout tile (16 rows each)
  const int halfB = (wv >> 2) * 64;                // w half (0 or 64)
  const int mlane = lane & 15;
  const int kb8   = (lane >> 4) << 3;              // A: K base 0 or 8
  const int kb16  = (lane >> 4) << 4;              // unused; B also uses kb8 (K base 0 or 8)

  const unsigned short* wSlabs =
      wprep + (size_t)blockIdx.y * 8 * 18432;      // this cout-block's 8 slabs
#if HAVE_TDM
  const unsigned wT_lds = (unsigned)(uintptr_t)(void*)&wT[0];
#endif

  v8f acc[4] = {};

  for (int cc = 0; cc < 8; ++cc) {
    const int cinBase = cc * 32;
    __syncthreads();                               // LDS free from prev iter

#if HAVE_TDM
    if (wv == 0) {
      tdm_load_weights(wT_lds,
                       (unsigned long long)(uintptr_t)(wSlabs + (size_t)cc * 18432));
    }
#else
    {
      const unsigned short* src = wSlabs + (size_t)cc * 18432;
      for (int i = tid; i < Kk * 64 * 32; i += 256) {
        int row = i >> 5, ci = i & 31;
        wT[row * XSTR + ci] = src[i];
      }
    }
#endif

    // stage shifted+padded x window: 32 ch x 136 w  (w0-4 .. w0+131)
    for (int i = tid; i < 32 * 136; i += 256) {
      int ci = i / 136;
      int wl = i - ci * 136;
      int wg = w0 - 4 + wl;
      float v = 0.f;
      if ((unsigned)wg < (unsigned)Ww) {
        int cin = cinBase + ci;
        int sw  = wg + sSh[cin];
        if (sw >= Ww) sw -= Ww;                    // circular shift
        v = x[(((size_t)b * Cc + cin) * Hh + h) * Ww + sw];
      }
      xT[wl * XSTR + ci] = f2bf(v);
    }

#if HAVE_TDM
    if (wv == 0) __builtin_amdgcn_s_wait_tensorcnt(0);
#endif
    __syncthreads();

#pragma unroll
    for (int k = 0; k < Kk; ++k) {
      const unsigned short* wr = &wT[(k * 64 + cm * 16 + mlane) * XSTR + kb8];
      V16 a;
      a.h[0] = *(const v8bf*)(const void*)(wr);
      a.h[1] = *(const v8bf*)(const void*)(wr + 16);
#pragma unroll
      for (int t = 0; t < 4; ++t) {
        const unsigned short* xr = &xT[(halfB + t * 16 + mlane + k) * XSTR + kb8];
        V16 bb;
        bb.h[0] = *(const v8bf*)(const void*)(xr);
        bb.h[1] = *(const v8bf*)(const void*)(xr + 16);
        acc[t] = __builtin_amdgcn_wmma_f32_16x16x32_bf16(
            false, a.v, false, bb.v, (short)0, acc[t], false, false);
      }
    }
  }

  // Epilogue: out = x + (1-gate)*relu(acc). D layout: VGPR v -> M = v (+8 for
  // lanes 16..31), N = lane&15.
  const int n  = lane & 15;
  const int hi = lane >> 4;
  for (int t = 0; t < 4; ++t) {
    const int wcol = w0 + halfB + t * 16 + n;
#pragma unroll
    for (int v = 0; v < 8; ++v) {
      const int m = cm * 16 + v + hi * 8;
      const size_t o = (((size_t)b * Cc + coutBase + m) * Hh + h) * Ww + wcol;
      float r = acc[t][v];
      r = r > 0.f ? r : 0.f;
      out[o] = x[o] + (1.0f - sGate[m]) * r;
    }
  }
}

extern "C" void kernel_launch(void* const* d_in, const int* in_sizes, int n_in,
                              void* d_out, int out_size, void* d_ws, size_t ws_size,
                              hipStream_t stream) {
  const float* x      = (const float*)d_in[0];
  const float* conv_w = (const float*)d_in[1];
  const float* fc_w   = (const float*)d_in[2];
  const float* fc_b   = (const float*)d_in[3];
  float* out = (float*)d_out;

  float* sq    = (float*)d_ws;                     // B*C floats
  float* gate  = sq + Bb * Cc;                     // B*C floats
  int*   shift = (int*)(gate + Bb * Cc);           // B*C ints
  unsigned short* wprep = (unsigned short*)(shift + Bb * Cc);  // 4*8*18432 bf16

  hsc_wprep_kernel<<<32, 256, 0, stream>>>(conv_w, wprep);
  hsc_mean_kernel<<<Bb * Cc, 256, 0, stream>>>(x, sq);
  hsc_gate_kernel<<<Bb, 256, 0, stream>>>(sq, fc_w, fc_b, gate, shift);
  hsc_conv_kernel<<<dim3(Ww / 128, Cc / 64, Bb * Hh), 256, 0, stream>>>(
      x, wprep, shift, gate, out);
}
